// RedditGATv2_73023033967281
// MI455X (gfx1250) — compile-verified, matrix operand order checked by reading, not run on hardware
//
#include <hip/hip_runtime.h>
#include <hip/hip_bf16.h>

// Problem constants (match reference)
#define NS 10000
#define NU 100000
#define NE 200000
#define DIN 384
#define HD 128
#define NH 5
#define NHD (NH*HD)   // 640

typedef __attribute__((ext_vector_type(16))) _Float16 v16h;
typedef __attribute__((ext_vector_type(8)))  _Float16 v8h;
typedef __attribute__((ext_vector_type(8)))  float    v8f;

static __device__ __forceinline__ float warp_sum(float v) {
    #pragma unroll
    for (int off = 16; off > 0; off >>= 1) v += __shfl_xor(v, off, 32);
    return v;
}

// ---------------- elementwise helpers ----------------

__global__ void fill_f32_kernel(float* p, float v, int n) {
    int i = blockIdx.x * blockDim.x + threadIdx.x;
    if (i < n) p[i] = v;
}
__global__ void fill_u32_kernel(unsigned* p, unsigned v, int n) {
    int i = blockIdx.x * blockDim.x + threadIdx.x;
    if (i < n) p[i] = v;
}
__global__ void cast_f16_kernel(const float* __restrict__ s, _Float16* __restrict__ d, int n) {
    int i = blockIdx.x * blockDim.x + threadIdx.x;
    if (i < n) d[i] = (_Float16)s[i];
}
// Wt[n*K+k] = W[k*N+n], cast to f16
__global__ void transpose_cast_kernel(const float* __restrict__ W, _Float16* __restrict__ Wt,
                                      int K, int N) {
    int i = blockIdx.x * blockDim.x + threadIdx.x;
    if (i >= K * N) return;
    int k = i / N, n = i - k * N;
    Wt[(size_t)n * K + k] = (_Float16)W[i];
}
// u1 = uacc + bias (in place), rows of 128
__global__ void add_bias_kernel(float* __restrict__ u, const float* __restrict__ bias, int n) {
    int i = blockIdx.x * blockDim.x + threadIdx.x;
    if (i < n) u[i] += bias[i & (HD - 1)];
}

// ---------------- WMMA GEMM: C[M,N] = A[M,K](f16) @ Bt[N,K](f16)^T + bias ----------------
// block = 256 threads = 8 waves arranged 4(M) x 2(N); per-wave tile = 16 rows x 64 cols
// (4 accumulators, A fragment reused across 4 WMMAs per K-step).
// Requires N % 128 == 0, K % 32 == 0. M ragged in 16s: waves past M exit uniformly
// (EXEC stays all-ones for live waves, as WMMA requires).
template<bool STORE_HALF>
__global__ void gemm_wmma_kernel(const _Float16* __restrict__ A, const _Float16* __restrict__ Bt,
                                 const float* __restrict__ bias, void* __restrict__ Cout,
                                 int M, int N, int K) {
    const int lane = threadIdx.x & 31;
    const int wave = threadIdx.x >> 5;
    const int wm = wave & 3;                    // 4 waves along M
    const int wn = wave >> 2;                   // 2 waves along N
    const int m0 = blockIdx.x * 64 + wm * 16;
    if (m0 >= M) return;                        // wave-uniform exit for partial last block
    const int n0 = blockIdx.y * 128 + wn * 64;
    const int mrow = m0 + (lane & 15);          // A fragment row
    const int nrow = n0 + (lane & 15);          // Bt row for N-tile 0
    const int kb = (lane >> 4) << 3;            // 0 or 8

    v8f acc[4] = {};
    for (int k0 = 0; k0 < K; k0 += 32) {
        const _Float16* ap = A + (size_t)mrow * K + k0 + kb;
        v8h alo = *(const v8h*)(ap);
        v8h ahi = *(const v8h*)(ap + 16);
        v16h a;
        #pragma unroll
        for (int i = 0; i < 8; ++i) { a[i] = alo[i]; a[i + 8] = ahi[i]; }
        #pragma unroll
        for (int t = 0; t < 4; ++t) {
            const _Float16* bp = Bt + (size_t)(nrow + t * 16) * K + k0 + kb;
            v8h blo = *(const v8h*)(bp);
            v8h bhi = *(const v8h*)(bp + 16);
            v16h b;
            #pragma unroll
            for (int i = 0; i < 8; ++i) { b[i] = blo[i]; b[i + 8] = bhi[i]; }
            acc[t] = __builtin_amdgcn_wmma_f32_16x16x32_f16(false, a, false, b,
                                                            (short)0, acc[t], false, false);
        }
    }
    // C/D layout: lanes 0-15 -> M = m0 + r ; lanes 16-31 -> M = m0 + 8 + r ; col = (lane&15)
    const int rbase = m0 + ((lane >> 4) << 3);
    #pragma unroll
    for (int t = 0; t < 4; ++t) {
        const int col = n0 + t * 16 + (lane & 15);
        const float bv = bias[col];
        #pragma unroll
        for (int r = 0; r < 8; ++r) {
            float v = acc[t][r] + bv;
            size_t off = (size_t)(rbase + r) * N + col;
            if (STORE_HALF) ((_Float16*)Cout)[off] = (_Float16)v;
            else            ((float*)Cout)[off]    = v;
        }
    }
}

// ---------------- row kernels (one wave per 128-wide row) ----------------

// LayerNorm row -> f16 (GEMM input prep)
__global__ void ln_rows_kernel(const float* __restrict__ x, const float* __restrict__ g,
                               const float* __restrict__ b, _Float16* __restrict__ out, int nrows) {
    int gw = (blockIdx.x * blockDim.x + threadIdx.x) >> 5;
    if (gw >= nrows) return;
    int lane = threadIdx.x & 31;
    const float* xr = x + (size_t)gw * HD;
    float v[4]; float s = 0.f;
    #pragma unroll
    for (int j = 0; j < 4; ++j) { v[j] = xr[lane + j * 32]; s += v[j]; }
    s = warp_sum(s);
    float mean = s * (1.0f / HD);
    float q = 0.f;
    #pragma unroll
    for (int j = 0; j < 4; ++j) { float d = v[j] - mean; q += d * d; }
    q = warp_sum(q);
    float inv = rsqrtf(q * (1.0f / HD) + 1e-5f);
    _Float16* o = out + (size_t)gw * HD;
    #pragma unroll
    for (int j = 0; j < 4; ++j) {
        int cx = lane + j * 32;
        o[cx] = (_Float16)((v[j] - mean) * inv * g[cx] + b[cx]);
    }
}

// LN -> LeakyReLU(0.2) -> L2 normalize -> f32 (NodeProjection epilogue)
__global__ void proj_post_kernel(const float* __restrict__ x, const float* __restrict__ g,
                                 const float* __restrict__ b, float* __restrict__ out, int nrows) {
    int gw = (blockIdx.x * blockDim.x + threadIdx.x) >> 5;
    if (gw >= nrows) return;
    int lane = threadIdx.x & 31;
    const float* xr = x + (size_t)gw * HD;
    float v[4]; float s = 0.f;
    #pragma unroll
    for (int j = 0; j < 4; ++j) { v[j] = xr[lane + j * 32]; s += v[j]; }
    s = warp_sum(s);
    float mean = s * (1.0f / HD);
    float q = 0.f;
    #pragma unroll
    for (int j = 0; j < 4; ++j) { float d = v[j] - mean; q += d * d; }
    q = warp_sum(q);
    float inv = rsqrtf(q * (1.0f / HD) + 1e-5f);
    float y[4]; float n2 = 0.f;
    #pragma unroll
    for (int j = 0; j < 4; ++j) {
        int cx = lane + j * 32;
        float t = (v[j] - mean) * inv * g[cx] + b[cx];
        t = (t >= 0.f) ? t : 0.2f * t;
        y[j] = t; n2 += t * t;
    }
    n2 = warp_sum(n2);
    float nrm = fmaxf(sqrtf(n2), 1e-12f);
    float* o = out + (size_t)gw * HD;
    #pragma unroll
    for (int j = 0; j < 4; ++j) o[lane + j * 32] = y[j] / nrm;
}

// (u2acc + bias + u1) -> L2 normalize -> out (final user rows)
__global__ void residual_l2_kernel(const float* __restrict__ uacc, const float* __restrict__ bias,
                                   const float* __restrict__ u1, float* __restrict__ out, int nrows) {
    int gw = (blockIdx.x * blockDim.x + threadIdx.x) >> 5;
    if (gw >= nrows) return;
    int lane = threadIdx.x & 31;
    size_t base = (size_t)gw * HD;
    float y[4]; float n2 = 0.f;
    #pragma unroll
    for (int j = 0; j < 4; ++j) {
        int cx = lane + j * 32;
        float t = uacc[base + cx] + bias[cx] + u1[base + cx];
        y[j] = t; n2 += t * t;
    }
    n2 = warp_sum(n2);
    float nrm = fmaxf(sqrtf(n2), 1e-12f);
    #pragma unroll
    for (int j = 0; j < 4; ++j) out[base + lane + j * 32] = y[j] / nrm;
}

// ---------------- edge kernels ----------------
// monotone float<->uint map for atomicMax on floats
static __device__ __forceinline__ unsigned f2o(float f) {
    unsigned u = __float_as_uint(f);
    return (u & 0x80000000u) ? ~u : (u | 0x80000000u);
}
static __device__ __forceinline__ float o2f(unsigned u) {
    return (u & 0x80000000u) ? __uint_as_float(u ^ 0x80000000u) : __uint_as_float(~u);
}

// one wave per (edge, head): alpha = sum_c att*lrelu(xl[src]+xr[dst]+ee); segment max via atomics
__global__ void edge_alpha_kernel(const _Float16* __restrict__ xl, const _Float16* __restrict__ xr,
                                  const float* __restrict__ ea, const int* __restrict__ src,
                                  const int* __restrict__ dst, const float* __restrict__ We,
                                  const float* __restrict__ att, float* __restrict__ alpha,
                                  unsigned* __restrict__ amax) {
    int gw = (blockIdx.x * blockDim.x + threadIdx.x) >> 5;
    if (gw >= NE * NH) return;
    int e = gw / NH, h = gw - e * NH;
    int lane = threadIdx.x & 31;
    int s = src[e], d = dst[e];
    float a0 = ea[e * 2 + 0], a1 = ea[e * 2 + 1];
    const _Float16* xls = xl + (size_t)s * NHD + h * HD;
    const _Float16* xrd = xr + (size_t)d * NHD + h * HD;
    const float* w0 = We + h * HD;
    const float* w1 = We + NHD + h * HD;
    const float* at = att + h * HD;
    float acc = 0.f;
    #pragma unroll
    for (int j = 0; j < 4; ++j) {
        int cx = lane + j * 32;
        float ee = a0 * w0[cx] + a1 * w1[cx];
        float m = (float)xls[cx] + (float)xrd[cx] + ee;
        m = (m >= 0.f) ? m : 0.2f * m;
        acc += m * at[cx];
    }
    acc = warp_sum(acc);
    if (lane == 0) {
        alpha[gw] = acc;
        atomicMax(amax + (size_t)d * NH + h, f2o(acc));
    }
}

// one thread per (edge, head): ex = exp(alpha - amax[dst]); den[dst] += ex
__global__ void edge_exp_kernel(float* __restrict__ alpha, const unsigned* __restrict__ amax,
                                const int* __restrict__ dst, float* __restrict__ den) {
    int i = blockIdx.x * blockDim.x + threadIdx.x;
    if (i >= NE * NH) return;
    int e = i / NH, h = i - e * NH;
    int d = dst[e];
    float mx = o2f(amax[(size_t)d * NH + h]);
    float ex = expf(alpha[i] - mx);
    alpha[i] = ex;
    atomicAdd(den + (size_t)d * NH + h, ex);
}

// one wave per (edge, head): uacc[dst,c] += (1/NH) * w * xl[src, h*128+c]
__global__ void edge_scatter_kernel(const float* __restrict__ ex, const float* __restrict__ den,
                                    const _Float16* __restrict__ xl, const int* __restrict__ src,
                                    const int* __restrict__ dst, float* __restrict__ uacc) {
    int gw = (blockIdx.x * blockDim.x + threadIdx.x) >> 5;
    if (gw >= NE * NH) return;
    int e = gw / NH, h = gw - e * NH;
    int lane = threadIdx.x & 31;
    int s = src[e], d = dst[e];
    float coef = (1.0f / NH) * ex[gw] / den[(size_t)d * NH + h];
    const _Float16* xls = xl + (size_t)s * NHD + h * HD;
    float* o = uacc + (size_t)d * HD;
    #pragma unroll
    for (int j = 0; j < 4; ++j) {
        int cx = lane + j * 32;
        atomicAdd(o + cx, coef * (float)xls[cx]);
    }
}

// ---------------- host orchestration ----------------

static inline int cdiv(long long a, long long b) { return (int)((a + b - 1) / b); }

extern "C" void kernel_launch(void* const* d_in, const int* in_sizes, int n_in,
                              void* d_out, int out_size, void* d_ws, size_t ws_size,
                              hipStream_t stream) {
    (void)in_sizes; (void)n_in; (void)out_size; (void)ws_size;
    // setup_inputs() flat order (params dict in insertion order)
    const float* x_sub  = (const float*)d_in[0];
    const float* x_user = (const float*)d_in[1];
    const float* eattr  = (const float*)d_in[2];
    const int*   src    = (const int*)  d_in[3];
    const int*   dst    = (const int*)  d_in[4];
    const float* sub_W  = (const float*)d_in[5];
    const float* sub_b  = (const float*)d_in[6];
    const float* sub_g  = (const float*)d_in[7];
    const float* sub_be = (const float*)d_in[8];
    const float* usr_W  = (const float*)d_in[9];
    const float* usr_b  = (const float*)d_in[10];
    const float* usr_g  = (const float*)d_in[11];
    const float* usr_be = (const float*)d_in[12];
    const float* ln_g[2]  = { (const float*)d_in[13], (const float*)d_in[22] };
    const float* ln_b[2]  = { (const float*)d_in[14], (const float*)d_in[23] };
    const float* Wl[2]    = { (const float*)d_in[15], (const float*)d_in[24] };
    const float* bl[2]    = { (const float*)d_in[16], (const float*)d_in[25] };
    const float* Wr[2]    = { (const float*)d_in[17], (const float*)d_in[26] };
    const float* br[2]    = { (const float*)d_in[18], (const float*)d_in[27] };
    const float* We[2]    = { (const float*)d_in[19], (const float*)d_in[28] };
    const float* att[2]   = { (const float*)d_in[20], (const float*)d_in[29] };
    const float* bias[2]  = { (const float*)d_in[21], (const float*)d_in[30] };
    float* out = (float*)d_out;

    // workspace bump allocator
    char* W = (char*)d_ws;
    size_t off = 0;
    auto alloc = [&](size_t bytes) { size_t p = off; off += (bytes + 255) & ~(size_t)255; return p; };
    _Float16* R0 = (_Float16*)(W + alloc((size_t)NU * NHD * 2)); // x_user f16, then xr f16
    _Float16* R1 = (_Float16*)(W + alloc((size_t)NS * NHD * 2)); // x_sub f16, then xl f16
    float*    R2 = (float*)   (W + alloc((size_t)NU * HD * 4));  // proj tmp, then u1
    float*    R3 = (float*)   (W + alloc((size_t)NU * HD * 4));  // usr proj, then u2 acc
    _Float16* R4 = (_Float16*)(W + alloc((size_t)NU * HD * 2));  // LN'd dst rows (f16)
    _Float16* R5 = (_Float16*)(W + alloc((size_t)NS * HD * 2));  // LN'd sub rows (f16)
    _Float16* R6 = (_Float16*)(W + alloc((size_t)NHD * DIN * 2));// transposed weight slot
    float*    R7 = (float*)   (W + alloc((size_t)NE * NH * 4));  // alpha / ex
    unsigned* R8 = (unsigned*)(W + alloc((size_t)NU * NH * 4));  // segment max (ordered uint)
    float*    R9 = (float*)   (W + alloc((size_t)NU * NH * 4));  // softmax denom

    const int TB = 256;
    const int EW_BLK = cdiv((long long)NE * NH * 32, TB); // wave-per-(edge,head) kernels
    const int ET_BLK = cdiv((long long)NE * NH, TB);      // thread-per-(edge,head)

    // ---- Phase A: node projections ----
    cast_f16_kernel<<<cdiv((long long)NS * DIN, TB), TB, 0, stream>>>(x_sub,  R1, NS * DIN);
    cast_f16_kernel<<<cdiv((long long)NU * DIN, TB), TB, 0, stream>>>(x_user, R0, NU * DIN);

    transpose_cast_kernel<<<cdiv(DIN * HD, TB), TB, 0, stream>>>(sub_W, R6, DIN, HD);
    gemm_wmma_kernel<false><<<dim3(cdiv(NS, 64), HD / 128), TB, 0, stream>>>(R1, R6, sub_b, R2, NS, HD, DIN);
    proj_post_kernel<<<cdiv(NS, 8), TB, 0, stream>>>(R2, sub_g, sub_be, out, NS); // sub -> d_out[0:NS]

    transpose_cast_kernel<<<cdiv(DIN * HD, TB), TB, 0, stream>>>(usr_W, R6, DIN, HD);
    gemm_wmma_kernel<false><<<dim3(cdiv(NU, 64), HD / 128), TB, 0, stream>>>(R0, R6, usr_b, R2, NU, HD, DIN);
    proj_post_kernel<<<cdiv(NU, 8), TB, 0, stream>>>(R2, usr_g, usr_be, R3, NU);  // usr -> R3

    // ---- Two GAT blocks ----
    for (int i = 0; i < 2; ++i) {
        const float* dstfeat = (i == 0) ? R3 : R2;   // block1: usr, block2: u1
        float*       uacc    = (i == 0) ? R2 : R3;   // block1: u1 acc, block2: u2 acc

        // xl = LN_i(sub) @ Wl + bl  -> R1 (f16 [NS,640])
        transpose_cast_kernel<<<cdiv(HD * NHD, TB), TB, 0, stream>>>(Wl[i], R6, HD, NHD);
        ln_rows_kernel<<<cdiv(NS, 8), TB, 0, stream>>>(out, ln_g[i], ln_b[i], R5, NS);
        gemm_wmma_kernel<true><<<dim3(cdiv(NS, 64), NHD / 128), TB, 0, stream>>>(R5, R6, bl[i], R1, NS, NHD, HD);

        // xr = LN_i(dstfeat) @ Wr + br -> R0 (f16 [NU,640])
        transpose_cast_kernel<<<cdiv(HD * NHD, TB), TB, 0, stream>>>(Wr[i], R6, HD, NHD);
        ln_rows_kernel<<<cdiv(NU, 8), TB, 0, stream>>>(dstfeat, ln_g[i], ln_b[i], R4, NU);
        gemm_wmma_kernel<true><<<dim3(cdiv(NU, 64), NHD / 128), TB, 0, stream>>>(R4, R6, br[i], R0, NU, NHD, HD);

        // attention logits + segment softmax + scatter-mean
        fill_u32_kernel<<<cdiv(NU * NH, TB), TB, 0, stream>>>(R8, 0x007FFFFFu, NU * NH); // map(-inf)
        edge_alpha_kernel<<<EW_BLK, TB, 0, stream>>>(R1, R0, eattr, src, dst, We[i], att[i], R7, R8);
        fill_f32_kernel<<<cdiv(NU * NH, TB), TB, 0, stream>>>(R9, 0.f, NU * NH);
        edge_exp_kernel<<<ET_BLK, TB, 0, stream>>>(R7, R8, dst, R9);
        fill_f32_kernel<<<cdiv((long long)NU * HD, TB), TB, 0, stream>>>(uacc, 0.f, NU * HD);
        edge_scatter_kernel<<<EW_BLK, TB, 0, stream>>>(R7, R9, R1, src, dst, uacc);

        if (i == 0) {
            // u1 = uacc + bias1 (in place in R2)
            add_bias_kernel<<<cdiv((long long)NU * HD, TB), TB, 0, stream>>>(R2, bias[0], NU * HD);
        } else {
            // user_out = l2(uacc2 + bias2 + u1) -> d_out[NS:]
            residual_l2_kernel<<<cdiv(NU, 8), TB, 0, stream>>>(R3, bias[1], R2, out + (size_t)NS * HD, NU);
        }
    }
}
